// Attention_76055280878233
// MI455X (gfx1250) — compile-verified, matrix operand order checked by reading, not run on hardware
//
#include <hip/hip_runtime.h>
#include <hip/hip_bf16.h>
#include <math.h>
#include <stdint.h>

#define S 1024
#define D 64
#define NHEADS 64          // B*H = 8*8
#define NTOT (S * S)       // 1M score elements per head
#define NB (1 << 18)       // rank histogram buckets (top 18 key bits)
#define HISHIFT 14         // 32 - 18

typedef __attribute__((ext_vector_type(16))) __bf16 v16bf;
typedef __attribute__((ext_vector_type(8)))  float  v8f;

#if defined(__has_builtin)
#if __has_builtin(__builtin_amdgcn_global_load_async_to_lds_b128)
#define HAVE_ASYNC_LDS 1
typedef int v4i_vec __attribute__((vector_size(16)));
typedef __attribute__((address_space(1))) v4i_vec as1_v4i;   // global int32x4
typedef __attribute__((address_space(3))) v4i_vec as3_v4i;   // LDS int32x4
#endif
#endif

// ---------- helpers ----------

// order-preserving float -> u32 (ascending)
__device__ __forceinline__ unsigned mono_key(float f) {
  unsigned u = __float_as_uint(f);
  return (u & 0x80000000u) ? ~u : (u | 0x80000000u);
}

// A/B fragment (16x32 bf16, row-major source, contiguous K).
// CDNA5 16-bit A layout: lanes 0-15 = rows 0-15 (K runs 0-7,16-23),
// lanes 16-31 = same rows (K runs 8-15,24-31). B symmetric with col=lane.
__device__ __forceinline__ v16bf frag_row(const float* base, int ld, int row,
                                          int k0, int hi) {
  const float* p0 = base + (size_t)row * ld + k0 + (hi ? 8 : 0);
  const float* p1 = p0 + 16;
  v16bf f;
#pragma unroll
  for (int e = 0; e < 8; ++e) {
    f[e]     = (__bf16)p0[e];
    f[e + 8] = (__bf16)p1[e];
  }
  return f;
}

// B fragment when K runs down a strided (column-major w.r.t. K) source.
__device__ __forceinline__ v16bf frag_col(const float* base, int ld, int col,
                                          int k0, int hi) {
  int ka = k0 + (hi ? 8 : 0);
  v16bf f;
#pragma unroll
  for (int e = 0; e < 8; ++e) {
    f[e]     = (__bf16)base[(size_t)(ka + e) * ld + col];
    f[e + 8] = (__bf16)base[(size_t)(ka + 16 + e) * ld + col];
  }
  return f;
}

// ---------- kernels ----------

// Reciprocal row L2 norms 1/(|row|+EPS) of Q and K for all heads.
__global__ __launch_bounds__(256) void norms_kernel(const float* __restrict__ Q,
                                                    const float* __restrict__ K,
                                                    float* __restrict__ rq,
                                                    float* __restrict__ rk) {
  int gr   = blockIdx.x * 8 + (threadIdx.x >> 5);
  int lane = threadIdx.x & 31;
  const float* src;
  float* dst;
  int r;
  if (gr < NHEADS * S) { src = Q; dst = rq; r = gr; }
  else                 { src = K; dst = rk; r = gr - NHEADS * S; }
  float a = src[(size_t)r * D + lane];
  float b = src[(size_t)r * D + lane + 32];
  float s = a * a + b * b;
#pragma unroll
  for (int off = 16; off; off >>= 1) s += __shfl_down(s, off, 32);
  if (lane == 0) dst[r] = __builtin_amdgcn_rcpf(sqrtf(s) + 1e-5f);
}

// score = (Q . K^T) * rq[m] * rk[n] for one head, bf16 WMMA.
// Block = 8 waves sharing one 16x64 Q tile (staged in LDS via async copy);
// each wave computes one 16x16 C tile and fuses the rank histogram.
__global__ __launch_bounds__(256) void gemm_qk(const float* __restrict__ Q,
                                               const float* __restrict__ K,
                                               const float* __restrict__ rq,
                                               const float* __restrict__ rk,
                                               float* __restrict__ score,
                                               unsigned* __restrict__ cnt) {
  __shared__ __align__(16) float qtile[16 * 64];   // shared A tile (4 KB)

  int wave = threadIdx.x >> 5, lane = threadIdx.x & 31;
  int hi = lane >> 4, ln = lane & 15;
  int mBase = (blockIdx.x >> 3) * 16;                       // shared across block
  int n0 = (((blockIdx.x & 7) << 3) + wave) << 4;           // per-wave

  // Cooperative stage of Q[mBase..mBase+15][0..63]: 256 thr x 16 B.
  {
    int t = threadIdx.x;
    const float* gsrc = Q + (size_t)(mBase + (t >> 4)) * D + ((t & 15) << 2);
    float* ldst = &qtile[((t >> 4) << 6) + ((t & 15) << 2)];
#ifdef HAVE_ASYNC_LDS
    __builtin_amdgcn_global_load_async_to_lds_b128(
        (as1_v4i*)(uintptr_t)gsrc, (as3_v4i*)(uint32_t)(uintptr_t)ldst, 0, 0);
#if __has_builtin(__builtin_amdgcn_s_wait_asynccnt)
    __builtin_amdgcn_s_wait_asynccnt(0);
#else
    asm volatile("s_wait_asynccnt 0" ::: "memory");
#endif
#else
    *(float4*)ldst = *(const float4*)gsrc;
#endif
  }
  __syncthreads();

  v8f acc = {};
#pragma unroll
  for (int kt = 0; kt < D; kt += 32) {
    v16bf a = frag_row(qtile, 64, ln, kt, hi);     // ds_load from staged tile
    v16bf b = frag_row(K, D, n0 + ln, kt, hi);     // B col n = K row n (contiguous)
    acc = __builtin_amdgcn_wmma_f32_16x16x32_bf16(false, a, false, b,
                                                  (short)0, acc, false, false);
  }

  int n = n0 + ln;
  float rkn = rk[n];
#pragma unroll
  for (int r = 0; r < 8; ++r) {
    int m = mBase + r + hi * 8;                    // C layout: VGPR r -> row r (+8 hi)
    float c = acc[r] * (rq[m] * rkn);
    score[(size_t)m * S + n] = c;
    atomicAdd(&cnt[mono_key(c) >> HISHIFT], 1u);   // fused rank histogram
  }
}

__global__ __launch_bounds__(256) void zero_u32(unsigned* __restrict__ p, int n) {
  int i = blockIdx.x * 256 + threadIdx.x;
  if (i < n) p[i] = 0u;
}

// Descending exclusive prefix over buckets: startD[b] = #elems in buckets > b.
__global__ __launch_bounds__(1024) void scan_kernel(const unsigned* __restrict__ cnt,
                                                    unsigned* __restrict__ startD,
                                                    unsigned* __restrict__ wptr) {
  __shared__ unsigned sh[1024];
  unsigned running = 0;
  for (int chunk = 0; chunk < NB / 1024; ++chunk) {
    int j = chunk * 1024 + threadIdx.x;
    int b = NB - 1 - j;                     // walk buckets descending
    unsigned v = cnt[b];
    sh[threadIdx.x] = v;
    __syncthreads();
    for (int off = 1; off < 1024; off <<= 1) {
      unsigned t = (threadIdx.x >= (unsigned)off) ? sh[threadIdx.x - off] : 0u;
      __syncthreads();
      sh[threadIdx.x] += t;
      __syncthreads();
    }
    unsigned res = running + sh[threadIdx.x] - v;
    startD[b] = res;
    wptr[b]   = res;
    running += sh[1023];
    __syncthreads();
  }
}

// Scatter (key, idx) into bucket-sorted order; 4 elements per thread.
__global__ __launch_bounds__(256) void scatter_kernel(const float* __restrict__ score,
                                                      unsigned* __restrict__ wptr,
                                                      unsigned* __restrict__ skey,
                                                      unsigned* __restrict__ sidx) {
  int i4 = (blockIdx.x * 256 + threadIdx.x) << 2;
  float4 f = *(const float4*)(score + i4);
#pragma unroll
  for (int e = 0; e < 4; ++e) {
    float v = (e == 0) ? f.x : (e == 1) ? f.y : (e == 2) ? f.z : f.w;
    unsigned k = mono_key(v);
    unsigned pos = atomicAdd(&wptr[k >> HISHIFT], 1u);
    skey[pos] = k;
    sidx[pos] = (unsigned)(i4 + e);
  }
}

// Exact global rank via within-bucket counting; write unnormalized weight
// -log((rank+1)/N) back to the score buffer at the original flat index.
__global__ __launch_bounds__(256) void rank_kernel(const unsigned* __restrict__ skey,
                                                   const unsigned* __restrict__ sidx,
                                                   const unsigned* __restrict__ startD,
                                                   const unsigned* __restrict__ cnt,
                                                   float* __restrict__ weight) {
  int p = blockIdx.x * 256 + threadIdx.x;
  unsigned myK = skey[p], myI = sidx[p];
  unsigned b = myK >> HISHIFT;
  unsigned st = startD[b], en = st + cnt[b];
  unsigned r = 0;
  for (unsigned q = st; q < en; ++q) {
    unsigned k2 = skey[q];
    r += (k2 > myK) || (k2 == myK && sidx[q] < myI);  // stable descending ties
  }
  unsigned rank = st + r;
  // -log((rank+1)/2^20) = log(2^20) - log(rank+1)
  weight[myI] = 13.862943611198906f - logf((float)(rank + 1));
}

// Per-row reciprocal weight sums (deterministic tree reduction).
__global__ __launch_bounds__(256) void rowsum_kernel(const float* __restrict__ w,
                                                     float* __restrict__ rinv) {
  __shared__ float sh[256];
  int row = blockIdx.x;
  float s = 0.f;
  for (int c = threadIdx.x; c < S; c += 256) s += w[(size_t)row * S + c];
  sh[threadIdx.x] = s;
  __syncthreads();
  for (int off = 128; off; off >>= 1) {
    if ((int)threadIdx.x < off) sh[threadIdx.x] += sh[threadIdx.x + off];
    __syncthreads();
  }
  if (threadIdx.x == 0) rinv[row] = __builtin_amdgcn_rcpf(sh[0]);
}

// out[m,:] = rinv[m] * (W[m,:] . V) for one head, bf16 WMMA.
__global__ __launch_bounds__(256) void gemm_wv(const float* __restrict__ W,
                                               const float* __restrict__ V,
                                               const float* __restrict__ rinv,
                                               float* __restrict__ out) {
  int wave = threadIdx.x >> 5, lane = threadIdx.x & 31;
  int tile = blockIdx.x * 8 + wave;          // 256 tiles = 64 x 4
  int m0 = (tile >> 2) * 16, n0 = (tile & 3) * 16;
  int hi = lane >> 4, ln = lane & 15;
  v8f acc = {};
  for (int kt = 0; kt < S; kt += 32) {
    v16bf a = frag_row(W, S, m0 + ln, kt, hi);
    v16bf b = frag_col(V, D, n0 + ln, kt, hi);  // B col n = V[:, n], stride D
    acc = __builtin_amdgcn_wmma_f32_16x16x32_bf16(false, a, false, b,
                                                  (short)0, acc, false, false);
  }
#pragma unroll
  for (int r = 0; r < 8; ++r) {
    int m = m0 + r + hi * 8;
    out[(size_t)m * D + n0 + ln] = acc[r] * rinv[m];
  }
}

// ---------- launch ----------

extern "C" void kernel_launch(void* const* d_in, const int* in_sizes, int n_in,
                              void* d_out, int out_size, void* d_ws, size_t ws_size,
                              hipStream_t stream) {
  const float* Q = (const float*)d_in[0];
  const float* K = (const float*)d_in[1];
  const float* V = (const float*)d_in[2];
  float* out = (float*)d_out;

  // Workspace layout (~15.5 MB), reused serially per head.
  char* ws = (char*)d_ws;
  float*    score  = (float*)   (ws);                              // 4 MB
  unsigned* skey   = (unsigned*)(ws + ((size_t)4  << 20));         // 4 MB
  unsigned* sidx   = (unsigned*)(ws + ((size_t)8  << 20));         // 4 MB
  unsigned* cnt    = (unsigned*)(ws + ((size_t)12 << 20));         // 1 MB
  unsigned* startD = (unsigned*)(ws + ((size_t)13 << 20));         // 1 MB
  unsigned* wptr   = (unsigned*)(ws + ((size_t)14 << 20));         // 1 MB
  float*    rq     = (float*)   (ws + ((size_t)15 << 20));         // 256 KB
  float*    rk     = (float*)   (ws + ((size_t)15 << 20) + (1 << 18)); // 256 KB
  float*    rinv   = (float*)   (ws + ((size_t)15 << 20) + (2 << 18)); // 4 KB

  // All Q/K reciprocal row norms up front (131072 rows, 8 per block).
  norms_kernel<<<(2 * NHEADS * S) / 8, 256, 0, stream>>>(Q, K, rq, rk);

  for (int h = 0; h < NHEADS; ++h) {
    const float* Qh = Q + (size_t)h * S * D;
    const float* Kh = K + (size_t)h * S * D;
    const float* Vh = V + (size_t)h * S * D;
    float* outh = out + (size_t)h * S * D;

    zero_u32<<<NB / 256, 256, 0, stream>>>(cnt, NB);
    gemm_qk<<<512, 256, 0, stream>>>(Qh, Kh, rq + (size_t)h * S,
                                     rk + (size_t)h * S, score, cnt);
    scan_kernel<<<1, 1024, 0, stream>>>(cnt, startD, wptr);
    scatter_kernel<<<NTOT / 1024, 256, 0, stream>>>(score, wptr, skey, sidx);
    rank_kernel<<<NTOT / 256, 256, 0, stream>>>(skey, sidx, startD, cnt, score);
    rowsum_kernel<<<S, 256, 0, stream>>>(score, rinv);
    gemm_wv<<<32, 256, 0, stream>>>(score, Vh, rinv, outh);
  }
}
